// Model_71700184039765
// MI455X (gfx1250) — matrix operand order repaired and checked by reading, not compiled, not run on hardware
//
#include <hip/hip_runtime.h>
#include <hip/hip_bf16.h>

typedef __attribute__((ext_vector_type(16))) __bf16 v16bf;
typedef __attribute__((ext_vector_type(8)))  __bf16 v8bf;
typedef __attribute__((ext_vector_type(8)))  float  v8f;

#define N_NODES 17
#define TB      8              // batch items per workgroup
#define ROWS    (TB * N_NODES) // 136 real rows
#define ROWS_P  144            // padded to 9 tiles of 16
#define MT_CNT  9
#define SA      424            // bufA stride (bf16 elems): 848B rows, 16B aligned, bank-skewed
#define SB      328            // bufB stride: 656B rows, 16B aligned, bank-skewed

// LDS layout (bytes):
//   bufA : ROWS_P*SA*2 = 122112
//   bufB : ROWS_P*SB*2 =  94464
//   Asm  : 289*4, Asp : 289*4, rs_sm : 17*4, rs_sp : 17*4
#define LDS_BYTES (ROWS_P*SA*2 + ROWS_P*SB*2 + 289*4*2 + 17*4*2)

// ---------------------------------------------------------------------------
// Weight prep: W fp32 [din][dout]  ->  WT bf16 [doutP][dinP] (transposed, 0-pad)
// ---------------------------------------------------------------------------
__global__ void prep_weights(const float* __restrict__ W, __bf16* __restrict__ WT,
                             int din, int dout, int dinP, int doutP) {
    int i = blockIdx.x * blockDim.x + threadIdx.x;
    int tot = dinP * doutP;
    if (i >= tot) return;
    int n = i / dinP;
    int k = i - n * dinP;
    float v = (k < din && n < dout) ? W[(size_t)k * dout + n] : 0.0f;
    WT[i] = (__bf16)v;
}

// ---------------------------------------------------------------------------
// In-place adjacency mix on LDS activations: x[b,m,:] = sum_n A[m,n] x[b,n,:]
// Safe in-place: each thread owns one (batch, feature) column, caches 17 vals.
// ---------------------------------------------------------------------------
template<int DIN, int S>
__device__ __forceinline__ void mix_nodes(__bf16* __restrict__ buf,
                                          const float* __restrict__ A) {
    for (int p = threadIdx.x; p < TB * DIN; p += 256) {
        int b = p / DIN;
        int f = p - b * DIN;
        __bf16* base = buf + (size_t)(b * N_NODES) * S + f;
        float x[N_NODES];
        #pragma unroll
        for (int n = 0; n < N_NODES; ++n) x[n] = (float)base[(size_t)n * S];
        #pragma unroll
        for (int m = 0; m < N_NODES; ++m) {
            float acc = 0.0f;
            #pragma unroll
            for (int n = 0; n < N_NODES; ++n) acc += A[m * N_NODES + n] * x[n];
            base[(size_t)m * S] = (__bf16)acc;
        }
    }
}

// ---------------------------------------------------------------------------
// One fused layer GEMM: out = relu( in[ROWS_P x DINP] * W[DINP x DOUTP] + rs⊗b )
// bf16 inputs, f32 accumulate via v_wmma_f32_16x16x32_bf16.
// Each wave owns one N-tile (16 cols), 9 M accumulators, streams K.
// ---------------------------------------------------------------------------
template<int DINP, int DOUT, int DOUTP, int SIN, int SOUT>
__device__ __forceinline__ void gemm_layer(const __bf16* __restrict__ in,
                                           __bf16* __restrict__ outb,
                                           const __bf16* __restrict__ WT,
                                           const float* __restrict__ bias,
                                           const float* __restrict__ rs) {
    const int wave  = threadIdx.x >> 5;
    const int lane  = threadIdx.x & 31;
    const int lhalf = lane >> 4;   // 0: lanes 0-15, 1: lanes 16-31
    const int l16   = lane & 15;
    constexpr int NT = DOUTP / 16;
    constexpr int KT = DINP / 32;

    for (int nt = wave; nt < NT; nt += 8) {
        const int n = nt * 16 + l16;                 // output column for this lane
        const float bn = (n < DOUT) ? bias[n] : 0.0f;

        v8f acc[MT_CNT];
        #pragma unroll
        for (int mt = 0; mt < MT_CNT; ++mt) acc[mt] = (v8f){0,0,0,0,0,0,0,0};

        // B fragment source: lane covers column n, K-range [kt*32 + lhalf*16, +16)
        const __bf16* wrow = WT + (size_t)(nt * 16 + l16) * DINP + lhalf * 16;

        for (int kt = 0; kt < KT; ++kt) {
            if (kt + 1 < KT) __builtin_prefetch(wrow + (size_t)(kt + 1) * 32, 0, 1);
            v16bf bfrag = *(const v16bf*)(wrow + (size_t)kt * 32);

            #pragma unroll
            for (int mt = 0; mt < MT_CNT; ++mt) {
                // A fragment: row = mt*16 + l16; elems 0..15 at K=kt*32+lhalf*16+s
                // (contiguous, matching B fragment).
                const __bf16* arow = in + (size_t)(mt * 16 + l16) * SIN
                                        + kt * 32 + lhalf * 16;
                union { v16bf v; struct { v8bf lo; v8bf hi; } h; } af;
                af.h.lo = *(const v8bf*)(arow);
                af.h.hi = *(const v8bf*)(arow + 8);
                acc[mt] = __builtin_amdgcn_wmma_f32_16x16x32_bf16(
                    false, af.v, false, bfrag, (short)0, acc[mt], false, false);
            }
        }

        // Epilogue: + rs[node]*b[n], ReLU, bf16 store to LDS out buffer.
        #pragma unroll
        for (int mt = 0; mt < MT_CNT; ++mt) {
            #pragma unroll
            for (int j = 0; j < 8; ++j) {
                int r = mt * 16 + j + lhalf * 8;     // C/D layout: vgpr j -> M=j (+8 hi half)
                float v = acc[mt][j] + rs[r % N_NODES] * bn;
                v = v > 0.0f ? v : 0.0f;
                outb[(size_t)r * SOUT + n] = (__bf16)v;
            }
        }
    }
}

// ---------------------------------------------------------------------------
// Fully fused 6-layer graph autoencoder. One WG = 8 batch items, all
// activations stay in LDS as bf16; HBM touched only for H, weights, output.
// ---------------------------------------------------------------------------
__global__ __launch_bounds__(256, 1)
void fused_gae(const float* __restrict__ H,
               const float* __restrict__ Asm, const float* __restrict__ Asp,
               const __bf16* __restrict__ WT0, const __bf16* __restrict__ WT1,
               const __bf16* __restrict__ WT2, const __bf16* __restrict__ WT3,
               const __bf16* __restrict__ WT4, const __bf16* __restrict__ WT5,
               const float* __restrict__ b0, const float* __restrict__ b1,
               const float* __restrict__ b2, const float* __restrict__ b3,
               const float* __restrict__ b4, const float* __restrict__ b5,
               float* __restrict__ out) {
    extern __shared__ char smem[];
    __bf16* bufA  = (__bf16*)smem;                       // [ROWS_P][SA]
    __bf16* bufB  = bufA + (size_t)ROWS_P * SA;          // [ROWS_P][SB]
    float*  Asm_l = (float*)(bufB + (size_t)ROWS_P * SB);
    float*  Asp_l = Asm_l + 289;
    float*  rs_sm = Asp_l + 289;
    float*  rs_sp = rs_sm + N_NODES;

    const int tid  = threadIdx.x;
    const int bat0 = blockIdx.x * TB;

    // Stage adjacency matrices and input tile (fp32 -> bf16, K padded to 32).
    for (int i = tid; i < 289; i += 256) { Asm_l[i] = Asm[i]; Asp_l[i] = Asp[i]; }
    for (int i = tid; i < ROWS_P * 32; i += 256) {
        int r = i >> 5, k = i & 31;
        float v = 0.0f;
        if (r < ROWS && k < 2) v = H[((size_t)(bat0 * N_NODES) + r) * 2 + k];
        bufB[(size_t)r * SB + k] = (__bf16)v;
    }
    __syncthreads();
    if (tid < N_NODES) {
        float s = 0.0f, t = 0.0f;
        for (int n = 0; n < N_NODES; ++n) {
            s += Asm_l[tid * N_NODES + n];
            t += Asp_l[tid * N_NODES + n];
        }
        rs_sm[tid] = s; rs_sp[tid] = t;
    }
    __syncthreads();

    // --- encoder (smoothing adjacency) ---
    mix_nodes<2,   SB>(bufB, Asm_l); __syncthreads();
    gemm_layer< 32, 400, 416, SB, SA>(bufB, bufA, WT0, b0, rs_sm); __syncthreads();

    mix_nodes<400, SA>(bufA, Asm_l); __syncthreads();
    gemm_layer<416, 300, 320, SA, SB>(bufA, bufB, WT1, b1, rs_sm); __syncthreads();

    mix_nodes<300, SB>(bufB, Asm_l); __syncthreads();
    gemm_layer<320, 100, 128, SB, SA>(bufB, bufA, WT2, b2, rs_sm); __syncthreads();

    // --- decoder (sharpening adjacency) ---
    mix_nodes<100, SA>(bufA, Asp_l); __syncthreads();
    gemm_layer<128, 300, 320, SA, SB>(bufA, bufB, WT3, b3, rs_sp); __syncthreads();

    mix_nodes<300, SB>(bufB, Asp_l); __syncthreads();
    gemm_layer<320, 400, 416, SB, SA>(bufB, bufA, WT4, b4, rs_sp); __syncthreads();

    mix_nodes<400, SA>(bufA, Asp_l); __syncthreads();
    gemm_layer<416,   2,  16, SA, SB>(bufA, bufB, WT5, b5, rs_sp); __syncthreads();

    // Final masked store: [B,17,2] fp32.
    for (int i = tid; i < ROWS * 2; i += 256) {
        int r = i >> 1, e = i & 1;
        out[((size_t)(bat0 * N_NODES) + r) * 2 + e] = (float)bufB[(size_t)r * SB + e];
    }
}

// ---------------------------------------------------------------------------
extern "C" void kernel_launch(void* const* d_in, const int* in_sizes, int n_in,
                              void* d_out, int out_size, void* d_ws, size_t ws_size,
                              hipStream_t stream) {
    (void)in_sizes; (void)n_in; (void)out_size; (void)ws_size;
    const float* H   = (const float*)d_in[0];
    const float* Asm = (const float*)d_in[1];
    const float* Asp = (const float*)d_in[2];
    const float* W[6]  = {(const float*)d_in[3],  (const float*)d_in[5],
                          (const float*)d_in[7],  (const float*)d_in[9],
                          (const float*)d_in[11], (const float*)d_in[13]};
    const float* Bb[6] = {(const float*)d_in[4],  (const float*)d_in[6],
                          (const float*)d_in[8],  (const float*)d_in[10],
                          (const float*)d_in[12], (const float*)d_in[14]};

    const int din[6]   = {  2, 400, 300, 100, 300, 400};
    const int dout[6]  = {400, 300, 100, 300, 400,   2};
    const int dinP[6]  = { 32, 416, 320, 128, 320, 416};
    const int doutP[6] = {416, 320, 128, 320, 416,  16};

    __bf16* wt = (__bf16*)d_ws;
    size_t off[7]; off[0] = 0;
    for (int l = 0; l < 6; ++l) off[l + 1] = off[l] + (size_t)dinP[l] * doutP[l];

    for (int l = 0; l < 6; ++l) {
        int tot = dinP[l] * doutP[l];
        prep_weights<<<(tot + 255) / 256, 256, 0, stream>>>(
            W[l], wt + off[l], din[l], dout[l], dinP[l], doutP[l]);
    }

    (void)hipFuncSetAttribute((const void*)fused_gae,
                              hipFuncAttributeMaxDynamicSharedMemorySize,
                              (int)LDS_BYTES);

    fused_gae<<<16384 / TB, 256, LDS_BYTES, stream>>>(
        H, Asm, Asp,
        wt + off[0], wt + off[1], wt + off[2], wt + off[3], wt + off[4], wt + off[5],
        Bb[0], Bb[1], Bb[2], Bb[3], Bb[4], Bb[5],
        (float*)d_out);
}
